// ParallelTransformerBlock_75359496175688
// MI455X (gfx1250) — compile-verified, matrix operand order checked by reading, not run on hardware
//
#include <hip/hip_runtime.h>
#include <hip/hip_bf16.h>

// ---------- CDNA5 WMMA types ----------
typedef __bf16        v16bf __attribute__((ext_vector_type(16)));
typedef float         v8f   __attribute__((ext_vector_type(8)));
typedef unsigned int  v4u   __attribute__((ext_vector_type(4)));

union Frag {
    v16bf v;
    v4u   q[2];
    unsigned int   u[8];
    unsigned short h[16];
};

__device__ inline unsigned short f2bf(float f) {
    unsigned int x = __float_as_uint(f);
    unsigned int lsb = (x >> 16) & 1u;
    x += 0x7fffu + lsb;                 // round-to-nearest-even
    return (unsigned short)(x >> 16);
}
__device__ inline float bf2f(unsigned short h) {
    return __uint_as_float((unsigned int)h << 16);
}

__device__ inline v8f wmma_bf16(const Frag& a, const Frag& b, v8f c) {
    return __builtin_amdgcn_wmma_f32_16x16x32_bf16(
        false, a.v, false, b.v, (short)0, c, false, false);
}

// ---------- constants ----------
#define DIMC   2048
#define NPROJ  18688     // 2048 + 128 + 128 + 16384
#define FFI    8192
#define HEADS  16
#define DHEAD  128
#define SEQ    2048
#define TOK    4096      // B * SEQ

// ============================================================
// LayerNorm + cast to bf16:  xnb[row][j]
// ============================================================
__global__ __launch_bounds__(256)
void ln_kernel(const float* __restrict__ x, const float* __restrict__ gamma,
               const float* __restrict__ beta, unsigned short* __restrict__ xnb) {
    int row = blockIdx.x;
    int tid = threadIdx.x;
    const float* xr = x + (size_t)row * DIMC;
    float s = 0.f, s2 = 0.f;
    for (int j = tid; j < DIMC; j += 256) {
        float v = xr[j];
        s += v; s2 += v * v;
    }
    __shared__ float rs[256];
    __shared__ float rq[256];
    rs[tid] = s; rq[tid] = s2;
    __syncthreads();
    for (int st = 128; st > 0; st >>= 1) {
        if (tid < st) { rs[tid] += rs[tid + st]; rq[tid] += rq[tid + st]; }
        __syncthreads();
    }
    float mean = rs[0] * (1.0f / DIMC);
    float var  = rq[0] * (1.0f / DIMC) - mean * mean;
    float inv  = rsqrtf(var + 1e-5f);
    unsigned short* orow = xnb + (size_t)row * DIMC;
    for (int j = tid; j < DIMC; j += 256)
        orow[j] = f2bf((xr[j] - mean) * inv * gamma[j] + beta[j]);
}

// ============================================================
// fp32 -> bf16 elementwise cast (weights)
// ============================================================
__global__ __launch_bounds__(256)
void cast_bf16(const float* __restrict__ src, unsigned short* __restrict__ dst, size_t n) {
    for (size_t i = (size_t)blockIdx.x * blockDim.x + threadIdx.x; i < n;
         i += (size_t)gridDim.x * blockDim.x)
        dst[i] = f2bf(src[i]);
}

// ============================================================
// Tiled WMMA GEMM:  C[M,N] = A[M,K](bf16) * B[N,K](bf16)^T
// 128x128 tile, TK=32, 256 threads = 8 waves, wave = 32x64 sub-tile.
// M,N multiples of 128; K multiple of 32.
// mode: 0 = store f32, 1 = accumulate f32, 2 = store bf16
// ============================================================
__global__ __launch_bounds__(256)
void gemm_bf16(const unsigned short* __restrict__ A, const unsigned short* __restrict__ Bm,
               void* __restrict__ Cout, int M, int N, int K, int mode) {
    __shared__ unsigned short As[128][32];
    __shared__ unsigned short Bs[128][32];
    int bm = blockIdx.y * 128;
    int bn = blockIdx.x * 128;
    int tid  = threadIdx.x;
    int wave = tid >> 5;
    int lane = tid & 31;
    int wm = (wave & 3) * 32;    // wave M offset
    int wn = (wave >> 2) * 64;   // wave N offset
    int l16   = lane & 15;
    int khalf = (lane >> 4) * 8; // ISA 16-bit A/B layout: K-half by lane/16

    v8f acc[2][4];
    for (int mi = 0; mi < 2; ++mi)
        for (int ni = 0; ni < 4; ++ni)
            for (int r = 0; r < 8; ++r) acc[mi][ni][r] = 0.f;

    for (int k0 = 0; k0 < K; k0 += 32) {
        // 128x32 bf16 tiles = 512 x 16B chunks, 2 per thread
        for (int i = 0; i < 2; ++i) {
            int chunk = tid + i * 256;
            int r = chunk >> 2;
            int c = (chunk & 3) * 8;
            *(v4u*)&As[r][c] = *(const v4u*)&A [(size_t)(bm + r) * K + k0 + c];
            *(v4u*)&Bs[r][c] = *(const v4u*)&Bm[(size_t)(bn + r) * K + k0 + c];
        }
        // explicit GL2 prefetch two K-tiles ahead (global_prefetch_b8)
        if (k0 + 64 < K) {
            int pr = tid >> 1;
            __builtin_prefetch(&A [(size_t)(bm + pr) * K + k0 + 64], 0, 0);
            __builtin_prefetch(&Bm[(size_t)(bn + pr) * K + k0 + 64], 0, 0);
        }
        __syncthreads();

        Frag af[2], bf[4];
        for (int mi = 0; mi < 2; ++mi) {
            const unsigned short* p = &As[wm + mi * 16 + l16][khalf];
            af[mi].q[0] = *(const v4u*)p;
            af[mi].q[1] = *(const v4u*)(p + 16);
        }
        for (int ni = 0; ni < 4; ++ni) {
            const unsigned short* p = &Bs[wn + ni * 16 + l16][khalf];
            bf[ni].q[0] = *(const v4u*)p;
            bf[ni].q[1] = *(const v4u*)(p + 16);
        }
        for (int mi = 0; mi < 2; ++mi)
            for (int ni = 0; ni < 4; ++ni)
                acc[mi][ni] = wmma_bf16(af[mi], bf[ni], acc[mi][ni]);
        __syncthreads();
    }

    // C/D layout: M = r + 8*(lane>=16), N = lane&15
    int rofs = (lane >> 4) * 8;
    for (int mi = 0; mi < 2; ++mi)
        for (int ni = 0; ni < 4; ++ni)
            for (int r = 0; r < 8; ++r) {
                int row = bm + wm + mi * 16 + rofs + r;
                int col = bn + wn + ni * 16 + l16;
                size_t idx = (size_t)row * N + col;
                float v = acc[mi][ni][r];
                if (mode == 2) {
                    ((unsigned short*)Cout)[idx] = f2bf(v);
                } else {
                    float* C = (float*)Cout;
                    C[idx] = (mode == 1) ? (C[idx] + v) : v;
                }
            }
}

// ============================================================
// Q prep: l2norm over d=128 + RoPE, write bf16 [b,h,n,d]
// proj is bf16 [TOK, NPROJ]
// ============================================================
__global__ __launch_bounds__(128)
void prep_q(const unsigned short* __restrict__ proj, unsigned short* __restrict__ qb) {
    int token = blockIdx.x;       // b*SEQ + i
    int h     = blockIdx.y;
    int j     = threadIdx.x;
    float v = bf2f(proj[(size_t)token * NPROJ + h * DHEAD + j]);
    __shared__ float red[128];
    __shared__ float sv[128];
    red[j] = v * v;
    __syncthreads();
    for (int s = 64; s > 0; s >>= 1) { if (j < s) red[j] += red[j + s]; __syncthreads(); }
    float t = v / fmaxf(sqrtf(red[0]), 1e-12f);
    sv[j] = t;
    __syncthreads();
    int i = token & (SEQ - 1);
    int b = token >> 11;
    float ang = (float)i * __powf(10000.f, -(float)(j & 63) / 64.f);
    float rot = (j < 64) ? -sv[j + 64] : sv[j - 64];
    float o = t * __cosf(ang) + rot * __sinf(ang);
    qb[(((size_t)b * HEADS + h) * SEQ + i) * DHEAD + j] = f2bf(o);
}

// ============================================================
// K prep (l2norm + RoPE) and V cast, single kv head. [b,n,d]
// ============================================================
__global__ __launch_bounds__(128)
void prep_kv(const unsigned short* __restrict__ proj, unsigned short* __restrict__ kbuf,
             unsigned short* __restrict__ vbuf) {
    int token = blockIdx.x;
    int j     = threadIdx.x;
    const unsigned short* pr = proj + (size_t)token * NPROJ;
    float kv = bf2f(pr[2048 + j]);
    __shared__ float red[128];
    __shared__ float sv[128];
    red[j] = kv * kv;
    __syncthreads();
    for (int s = 64; s > 0; s >>= 1) { if (j < s) red[j] += red[j + s]; __syncthreads(); }
    float t = kv / fmaxf(sqrtf(red[0]), 1e-12f);
    sv[j] = t;
    __syncthreads();
    int i = token & (SEQ - 1);
    float ang = (float)i * __powf(10000.f, -(float)(j & 63) / 64.f);
    float rot = (j < 64) ? -sv[j + 64] : sv[j - 64];
    float o = t * __cosf(ang) + rot * __sinf(ang);
    kbuf[(size_t)token * DHEAD + j] = f2bf(o);
    vbuf[(size_t)token * DHEAD + j] = pr[2176 + j];   // bf16 passthrough
}

// ============================================================
// SwiGLU: act = silu(gate) * ff_x  -> bf16 [TOK, FFI]
// ============================================================
__global__ __launch_bounds__(256)
void ff_act(const unsigned short* __restrict__ proj, unsigned short* __restrict__ actb) {
    size_t total = (size_t)TOK * FFI;
    for (size_t idx = (size_t)blockIdx.x * blockDim.x + threadIdx.x; idx < total;
         idx += (size_t)gridDim.x * blockDim.x) {
        size_t token = idx >> 13;          // / 8192
        int    j     = (int)(idx & (FFI - 1));
        const unsigned short* pr = proj + token * NPROJ;
        float fx = bf2f(pr[2304 + j]);
        float g  = bf2f(pr[10496 + j]);
        float s  = g / (1.f + __expf(-g)); // silu
        actb[idx] = f2bf(s * fx);
    }
}

// ============================================================
// Flash attention (causal, MQA). 1 wave per (b,h, 16-query block).
// QK^T and P*V via v_wmma_f32_16x16x32_bf16, online softmax.
// Output bf16 [b,n,h*d].
// ============================================================
__global__ __launch_bounds__(32)
void attn_kernel(const unsigned short* __restrict__ qb, const unsigned short* __restrict__ kb,
                 const unsigned short* __restrict__ vb, unsigned short* __restrict__ ob) {
    int qblk = blockIdx.x;            // 0..SEQ/16-1
    int bh   = blockIdx.y;            // b*HEADS + h
    int b    = bh >> 4;
    int h    = bh & 15;
    int lane = threadIdx.x;
    int l16  = lane & 15;
    int half = lane >> 4;
    int khalf = half * 8;
    int q0 = qblk * 16;

    const unsigned short* Q  = qb + ((size_t)bh * SEQ + q0) * DHEAD;
    const unsigned short* Kp = kb + (size_t)b * SEQ * DHEAD;
    const unsigned short* Vp = vb + (size_t)b * SEQ * DHEAD;

    // Q fragments: 4 chunks of K=32 along d=128 (A layout)
    Frag qf[4];
    for (int kk = 0; kk < 4; ++kk) {
        const unsigned short* src = Q + (size_t)l16 * DHEAD + kk * 32;
        qf[kk].q[0] = *(const v4u*)(src + khalf);
        qf[kk].q[1] = *(const v4u*)(src + 16 + khalf);
    }

    v8f O[8];
    float m[8], l[8];
    for (int ni = 0; ni < 8; ++ni)
        for (int r = 0; r < 8; ++r) O[ni][r] = 0.f;
    for (int r = 0; r < 8; ++r) { m[r] = -1e30f; l[r] = 0.f; }

    __shared__ unsigned short Ps[16][32];

    int nkb = (q0 + 16 + 31) >> 5;    // key blocks of 32 (causal upper bound)
    for (int kbi = 0; kbi < nkb; ++kbi) {
        int kbase = kbi * 32;

        // sim tiles: two 16x16 tiles (keys kbase..+15, kbase+16..+31)
        v8f s0, s1;
        for (int r = 0; r < 8; ++r) { s0[r] = 0.f; s1[r] = 0.f; }
        for (int kk = 0; kk < 4; ++kk) {
            Frag kf0, kf1;
            const unsigned short* p0 = Kp + (size_t)(kbase + l16) * DHEAD + kk * 32;
            kf0.q[0] = *(const v4u*)(p0 + khalf);
            kf0.q[1] = *(const v4u*)(p0 + 16 + khalf);
            const unsigned short* p1 = Kp + (size_t)(kbase + 16 + l16) * DHEAD + kk * 32;
            kf1.q[0] = *(const v4u*)(p1 + khalf);
            kf1.q[1] = *(const v4u*)(p1 + 16 + khalf);
            s0 = wmma_bf16(qf[kk], kf0, s0);
            s1 = wmma_bf16(qf[kk], kf1, s1);
        }

        // scale + causal mask (C layout: row = q0 + r + 8*half, col key = kbase + l16 [+16])
        for (int r = 0; r < 8; ++r) {
            int qrow = q0 + r + half * 8;
            int key0 = kbase + l16;
            s0[r] = (key0      <= qrow) ? s0[r] * 8.0f : -1e30f;
            s1[r] = (key0 + 16 <= qrow) ? s1[r] * 8.0f : -1e30f;
        }

        // online softmax per row; reductions stay inside 16-lane halves
        for (int r = 0; r < 8; ++r) {
            float mx = fmaxf(s0[r], s1[r]);
            for (int off = 8; off >= 1; off >>= 1)
                mx = fmaxf(mx, __shfl_xor(mx, off, 32));
            float mnew = fmaxf(m[r], mx);
            float corr = __expf(m[r] - mnew);
            float p0 = __expf(s0[r] - mnew);
            float p1 = __expf(s1[r] - mnew);
            float sum = p0 + p1;
            for (int off = 8; off >= 1; off >>= 1)
                sum += __shfl_xor(sum, off, 32);
            l[r] = l[r] * corr + sum;
            m[r] = mnew;
            for (int ni = 0; ni < 8; ++ni) O[ni][r] *= corr;
            Ps[r + half * 8][l16]      = f2bf(p0);
            Ps[r + half * 8][16 + l16] = f2bf(p1);
        }
        __syncthreads();   // single wave: LDS in-order, barrier is cheap

        // P fragment (A layout 16x32) from LDS
        Frag pf;
        {
            const unsigned short* pr = &Ps[l16][0];
            pf.q[0] = *(const v4u*)(pr + khalf);
            pf.q[1] = *(const v4u*)(pr + 16 + khalf);
        }

        // O += P * V  (8 N-tiles of 16 over d=128), V gathered in B layout
        for (int ni = 0; ni < 8; ++ni) {
            Frag vf;
            int dcol = ni * 16 + l16;
            for (int e = 0; e < 8; ++e) {
                vf.h[e]     = Vp[(size_t)(kbase + khalf + e) * DHEAD + dcol];
                vf.h[e + 8] = Vp[(size_t)(kbase + 16 + khalf + e) * DHEAD + dcol];
            }
            O[ni] = wmma_bf16(pf, vf, O[ni]);
        }
        __syncthreads();
    }

    // epilogue: normalize, write bf16 [b, n, h*d]
    for (int ni = 0; ni < 8; ++ni)
        for (int r = 0; r < 8; ++r) {
            int row = q0 + r + half * 8;
            float o = O[ni][r] / l[r];
            ob[((size_t)b * SEQ + row) * (HEADS * DHEAD) + h * DHEAD + ni * 16 + l16] = f2bf(o);
        }
}

// ============================================================
// launch
// ============================================================
extern "C" void kernel_launch(void* const* d_in, const int* in_sizes, int n_in,
                              void* d_out, int out_size, void* d_ws, size_t ws_size,
                              hipStream_t stream) {
    (void)in_sizes; (void)n_in; (void)out_size; (void)ws_size;
    const float* x          = (const float*)d_in[0];
    const float* gamma      = (const float*)d_in[1];
    const float* beta       = (const float*)d_in[2];
    const float* w_fused    = (const float*)d_in[3];
    const float* w_attn_out = (const float*)d_in[4];
    const float* w_ff_out   = (const float*)d_in[5];
    float* out = (float*)d_out;

    char* ws = (char*)d_ws;
    size_t off = 0;
    auto alloc = [&](size_t bytes) {
        void* p = ws + off;
        off = (off + bytes + 255) & ~(size_t)255;
        return p;
    };
    unsigned short* xnb   = (unsigned short*)alloc((size_t)TOK * DIMC * 2);
    unsigned short* wfb   = (unsigned short*)alloc((size_t)NPROJ * DIMC * 2);
    unsigned short* waob  = (unsigned short*)alloc((size_t)DIMC * DIMC * 2);
    unsigned short* wffb  = (unsigned short*)alloc((size_t)DIMC * FFI * 2);
    unsigned short* proj  = (unsigned short*)alloc((size_t)TOK * NPROJ * 2);   // bf16 now
    unsigned short* qbuf  = (unsigned short*)alloc((size_t)TOK * HEADS * DHEAD * 2);
    unsigned short* kbuf  = (unsigned short*)alloc((size_t)TOK * DHEAD * 2);
    unsigned short* vbuf  = (unsigned short*)alloc((size_t)TOK * DHEAD * 2);
    unsigned short* attnb = (unsigned short*)alloc((size_t)TOK * DIMC * 2);
    unsigned short* actb  = (unsigned short*)alloc((size_t)TOK * FFI * 2);

    // 1) LayerNorm + bf16 cast
    ln_kernel<<<TOK, 256, 0, stream>>>(x, gamma, beta, xnb);

    // 2) weight casts to bf16
    cast_bf16<<<4096, 256, 0, stream>>>(w_fused,    wfb,  (size_t)NPROJ * DIMC);
    cast_bf16<<<2048, 256, 0, stream>>>(w_attn_out, waob, (size_t)DIMC * DIMC);
    cast_bf16<<<4096, 256, 0, stream>>>(w_ff_out,   wffb, (size_t)DIMC * FFI);

    // 3) fused projection GEMM: proj(bf16) = xn @ w_fused^T   (4096 x 18688, K=2048)
    {
        dim3 g(NPROJ / 128, TOK / 128);
        gemm_bf16<<<g, 256, 0, stream>>>(xnb, wfb, proj, TOK, NPROJ, DIMC, 2);
    }

    // 4) q/k/v prep + SwiGLU activation
    prep_q <<<dim3(TOK, HEADS), 128, 0, stream>>>(proj, qbuf);
    prep_kv<<<TOK, 128, 0, stream>>>(proj, kbuf, vbuf);
    ff_act <<<4096, 256, 0, stream>>>(proj, actb);

    // 5) flash attention
    attn_kernel<<<dim3(SEQ / 16, 2 * HEADS), 32, 0, stream>>>(qbuf, kbuf, vbuf, attnb);

    // 6) out = attn @ w_attn_out^T ; out += act @ w_ff_out^T
    {
        dim3 g(DIMC / 128, TOK / 128);
        gemm_bf16<<<g, 256, 0, stream>>>(attnb, waob, out, TOK, DIMC, DIMC, 0);
        gemm_bf16<<<g, 256, 0, stream>>>(actb,  wffb, out, TOK, DIMC, FFI,  1);
    }
}
